// LSTM_Model_bidirectional_75222057222483
// MI455X (gfx1250) — compile-verified, matrix operand order checked by reading, not run on hardware
//
#include <hip/hip_runtime.h>
#include <hip/hip_bf16.h>

// ---------------------------------------------------------------------------
// BiLSTM text classifier for MI455X (gfx1250, wave32).
//   text[64,1024] -> emb gather -> BiLSTM(H=10) x2 -> maxpool(T) -> MLP -> [64,4]
//
// Strategy:
//  * Input projections -> bf16 WMMA 16x16x32. Both directions' weights are
//    concatenated to N=80 = 5 exact 16-wide tiles; each wave carries 5
//    accumulators so one gathered/converted A fragment feeds 5 WMMAs
//    (5x less gather traffic + cvt work than one-tile-per-wave).
//  * Sequential LSTM scan -> (dir, batch-group) blocks of 16x40 threads,
//    Whh/h in LDS, c in registers, global_prefetch of next timestep gates.
//  * Pool + head: trivial reduction kernels.
// ---------------------------------------------------------------------------

typedef __attribute__((ext_vector_type(16))) __bf16 v16bf;
typedef __attribute__((ext_vector_type(8)))  float  v8f;

#define EMBED 256
#define HID   10
#define GATES 40          // 4*HID, PyTorch gate order i,f,g,o
#define NCAT  80          // 2 directions * 40 gates (5 x 16 WMMA N-tiles)
#define NTILE 5
#define BATCH 64
#define TSEQ  1024
#define ROWS  (BATCH * TSEQ)   // 65536 = 4096 x 16 WMMA M-tiles

// --------------------------- weight preparation -----------------------------
// Pack Wih0 [2,40,256] -> bf16 [80,256]; Wih1 [2,40,20] -> bf16 [80,32] (K padded);
// biases b0,b1 -> float biascat[160].
__global__ void prep_weights(const float* __restrict__ Wih0, const float* __restrict__ b0,
                             const float* __restrict__ Wih1, const float* __restrict__ b1,
                             __bf16* __restrict__ W0, __bf16* __restrict__ W1,
                             float* __restrict__ biascat) {
  int i = blockIdx.x * blockDim.x + threadIdx.x;
  if (i < NCAT * EMBED) {
    W0[i] = (__bf16)Wih0[i];                       // [2,40,256] is contiguous [80,256]
  }
  int j = i - NCAT * EMBED;
  if (j >= 0 && j < NCAT * 32) {
    int n = j / 32, k = j % 32;
    W1[j] = (k < 2 * HID) ? (__bf16)Wih1[n * (2 * HID) + k] : (__bf16)0.0f;
  }
  int m = j - NCAT * 32;
  if (m >= 0 && m < NCAT)           biascat[m] = b0[m];
  if (m >= NCAT && m < 2 * NCAT)    biascat[m] = b1[m - NCAT];
}

// ------------------- layer-0 input projection: gather + WMMA ----------------
// xW[row][n] = bias[n] + sum_k emb[text[row]][k] * W0[n][k]
// One wave per 16-row block; 5 accumulators cover all N=80 so the gathered
// A fragment (f32 -> bf16) is loaded/converted once per K-step.
__global__ void __launch_bounds__(32) gemm_xw0(const int* __restrict__ text,
                                               const float* __restrict__ emb,
                                               const __bf16* __restrict__ W0,
                                               const float* __restrict__ biascat,
                                               float* __restrict__ xW) {
  const int lane  = threadIdx.x;
  const int mt    = blockIdx.x;            // 0..4095
  const int nloc  = lane & 15;
  const int khalf = (lane >> 4) << 4;      // 0 or 16 (contiguous k order, same for A and B)
  const int row   = mt * 16 + nloc;

  const float* arow = emb + (size_t)text[row] * EMBED;   // gathered, L2-resident

  v8f acc[NTILE];
#pragma unroll
  for (int nt = 0; nt < NTILE; ++nt) {
    const float bias = biascat[nt * 16 + nloc];
#pragma unroll
    for (int r = 0; r < 8; ++r) acc[nt][r] = bias;
  }

#pragma unroll
  for (int k0 = 0; k0 < EMBED; k0 += 32) {
    v16bf a;
#pragma unroll
    for (int jj = 0; jj < 16; ++jj) a[jj] = (__bf16)arow[k0 + khalf + jj];
#pragma unroll
    for (int nt = 0; nt < NTILE; ++nt) {
      const v16bf b =
          *(const v16bf*)(W0 + (size_t)(nt * 16 + nloc) * EMBED + k0 + khalf);
      acc[nt] = __builtin_amdgcn_wmma_f32_16x16x32_bf16(false, a, false, b,
                                                        (short)0, acc[nt],
                                                        false, false);
    }
  }

  const int mhi = (lane >> 4) * 8;
#pragma unroll
  for (int nt = 0; nt < NTILE; ++nt)
#pragma unroll
    for (int r = 0; r < 8; ++r)
      xW[(size_t)(mt * 16 + mhi + r) * NCAT + nt * 16 + nloc] = acc[nt][r];
}

// ------------------- layer-1 input projection: WMMA, K=20 (padded 32) -------
__global__ void __launch_bounds__(32) gemm_xw1(const float* __restrict__ h0,
                                               const __bf16* __restrict__ W1,
                                               const float* __restrict__ biascat,
                                               float* __restrict__ xW) {
  const int lane  = threadIdx.x;
  const int mt    = blockIdx.x;
  const int nloc  = lane & 15;
  const int khalf = (lane >> 4) << 4;
  const int row   = mt * 16 + nloc;

  const float* arow = h0 + (size_t)row * (2 * HID);
  v16bf a;
#pragma unroll
  for (int jj = 0; jj < 16; ++jj) {
    int k = khalf + jj;
    a[jj] = (k < 2 * HID) ? (__bf16)arow[k] : (__bf16)0.0f;
  }

  v8f acc[NTILE];
#pragma unroll
  for (int nt = 0; nt < NTILE; ++nt) {
    const float bias = biascat[NCAT + nt * 16 + nloc];
#pragma unroll
    for (int r = 0; r < 8; ++r) acc[nt][r] = bias;
    const v16bf b = *(const v16bf*)(W1 + (size_t)(nt * 16 + nloc) * 32 + khalf);
    acc[nt] = __builtin_amdgcn_wmma_f32_16x16x32_bf16(false, a, false, b,
                                                      (short)0, acc[nt],
                                                      false, false);
  }

  const int mhi = (lane >> 4) * 8;
#pragma unroll
  for (int nt = 0; nt < NTILE; ++nt)
#pragma unroll
    for (int r = 0; r < 8; ++r)
      xW[(size_t)(mt * 16 + mhi + r) * NCAT + nt * 16 + nloc] = acc[nt][r];
}

// --------------------------- sequential LSTM scan ---------------------------
// grid.x = 8: dir = bx>>2, batch-group = bx&3 (16 samples each).
// block = 640 threads = 16 batch x 40 gates. Latency-bound recurrence:
// h broadcast via LDS, c in registers of the j<HID threads, next-step gate
// row prefetched (global_prefetch_b8).
__global__ void __launch_bounds__(640) lstm_scan(const float* __restrict__ xW,
                                                 const float* __restrict__ Whh,
                                                 float* __restrict__ hout) {
  const int dir = blockIdx.x >> 2;
  const int grp = blockIdx.x & 3;
  const int tid = threadIdx.x;
  const int bb  = tid / GATES;       // 0..15  local batch
  const int j   = tid % GATES;       // 0..39  gate index
  const int b   = grp * 16 + bb;

  __shared__ float whh[GATES][HID];
  __shared__ float hsh[16][HID];
  __shared__ float gsh[16][GATES];

  if (tid < GATES * HID) whh[tid / HID][tid % HID] = Whh[dir * GATES * HID + tid];
  if (tid < 16 * HID)    hsh[tid / HID][tid % HID] = 0.0f;
  float creg = 0.0f;                 // cell state, valid on j < HID threads
  __syncthreads();

  const float* xb = xW + (size_t)b * TSEQ * NCAT + dir * GATES;
  float*       hb = hout + (size_t)b * TSEQ * (2 * HID) + dir * HID;

  for (int it = 0; it < TSEQ; ++it) {
    const int t = dir ? (TSEQ - 1 - it) : it;
    if (it + 1 < TSEQ) {
      const int tn = dir ? (TSEQ - 2 - it) : (it + 1);
      __builtin_prefetch(xb + (size_t)tn * NCAT, 0, 0);   // global_prefetch_b8
    }

    float g = xb[(size_t)t * NCAT + j];
#pragma unroll
    for (int k = 0; k < HID; ++k) g += whh[j][k] * hsh[bb][k];
    gsh[bb][j] = g;
    __syncthreads();

    if (j < HID) {
      const float gi = gsh[bb][j];
      const float gf = gsh[bb][HID + j];
      const float gg = gsh[bb][2 * HID + j];
      const float go = gsh[bb][3 * HID + j];
      const float si = 1.0f / (1.0f + __expf(-gi));
      const float sf = 1.0f / (1.0f + __expf(-gf));
      const float so = 1.0f / (1.0f + __expf(-go));
      creg = sf * creg + si * tanhf(gg);
      const float h = so * tanhf(creg);
      hsh[bb][j] = h;
      hb[(size_t)t * (2 * HID) + j] = h;
    }
    __syncthreads();
  }
}

// ------------------------------- max pool over T ----------------------------
__global__ void __launch_bounds__(128) pool_max(const float* __restrict__ h1,
                                                float* __restrict__ pooled) {
  const int b = blockIdx.x, tid = threadIdx.x;
  float m[2 * HID];
#pragma unroll
  for (int c = 0; c < 2 * HID; ++c) m[c] = -3.402823466e38f;
  for (int t = tid; t < TSEQ; t += 128) {
    const float* r = h1 + ((size_t)b * TSEQ + t) * (2 * HID);
#pragma unroll
    for (int c = 0; c < 2 * HID; ++c) m[c] = fmaxf(m[c], r[c]);
  }
  __shared__ float red[128][2 * HID];
#pragma unroll
  for (int c = 0; c < 2 * HID; ++c) red[tid][c] = m[c];
  __syncthreads();
  for (int s = 64; s > 0; s >>= 1) {
    if (tid < s)
#pragma unroll
      for (int c = 0; c < 2 * HID; ++c)
        red[tid][c] = fmaxf(red[tid][c], red[tid + s][c]);
    __syncthreads();
  }
  if (tid < 2 * HID) pooled[b * (2 * HID) + tid] = red[0][tid];
}

// ---------------------------------- MLP head --------------------------------
__global__ void __launch_bounds__(64) head(const float* __restrict__ pooled,
                                           const float* __restrict__ fcw,
                                           const float* __restrict__ fcb,
                                           const float* __restrict__ fc2w,
                                           const float* __restrict__ fc2b,
                                           float* __restrict__ out) {
  const int b = threadIdx.x;
  if (b >= BATCH) return;
  float p[2 * HID];
#pragma unroll
  for (int c = 0; c < 2 * HID; ++c) p[c] = pooled[b * (2 * HID) + c];
  float z[HID];
#pragma unroll
  for (int h = 0; h < HID; ++h) {
    float s = fcb[h];
#pragma unroll
    for (int c = 0; c < 2 * HID; ++c) s += fcw[h * (2 * HID) + c] * p[c];
    z[h] = fmaxf(s, 0.0f);
  }
#pragma unroll
  for (int cl = 0; cl < 4; ++cl) {
    float s = fc2b[cl];
#pragma unroll
    for (int h = 0; h < HID; ++h) s += fc2w[cl * HID + h] * z[h];
    out[b * 4 + cl] = s;
  }
}

// --------------------------------- launcher ---------------------------------
extern "C" void kernel_launch(void* const* d_in, const int* in_sizes, int n_in,
                              void* d_out, int out_size, void* d_ws, size_t ws_size,
                              hipStream_t stream) {
  const int*   text = (const int*)d_in[0];
  const float* emb  = (const float*)d_in[1];
  const float* Wih0 = (const float*)d_in[2];
  const float* Whh0 = (const float*)d_in[3];
  const float* b0   = (const float*)d_in[4];
  const float* Wih1 = (const float*)d_in[5];
  const float* Whh1 = (const float*)d_in[6];
  const float* b1   = (const float*)d_in[7];
  const float* fcw  = (const float*)d_in[8];
  const float* fcb  = (const float*)d_in[9];
  const float* fc2w = (const float*)d_in[10];
  const float* fc2b = (const float*)d_in[11];
  float* out = (float*)d_out;

  // Workspace carve-up (floats). xW buffer is reused for layer 0 and layer 1.
  float* wsf     = (float*)d_ws;
  float* xW      = wsf;                                 // ROWS*NCAT  (21.0 MB)
  float* h0      = xW + (size_t)ROWS * NCAT;            // ROWS*20    ( 5.25 MB)
  float* h1      = h0 + (size_t)ROWS * 2 * HID;         // ROWS*20    ( 5.25 MB)
  float* pooled  = h1 + (size_t)ROWS * 2 * HID;         // 64*20
  float* biascat = pooled + BATCH * 2 * HID;            // 160
  __bf16* W0     = (__bf16*)(biascat + 2 * NCAT);       // 80*256 bf16 (32B aligned)
  __bf16* W1     = W0 + (size_t)NCAT * EMBED;           // 80*32 bf16

  const int prep_elems = NCAT * EMBED + NCAT * 32 + 2 * NCAT;
  prep_weights<<<(prep_elems + 255) / 256, 256, 0, stream>>>(Wih0, b0, Wih1, b1,
                                                             W0, W1, biascat);

  gemm_xw0<<<ROWS / 16, 32, 0, stream>>>(text, emb, W0, biascat, xW);
  lstm_scan<<<8, 640, 0, stream>>>(xW, Whh0, h0);
  gemm_xw1<<<ROWS / 16, 32, 0, stream>>>(h0, W1, biascat, xW);
  lstm_scan<<<8, 640, 0, stream>>>(xW, Whh1, h1);
  pool_max<<<BATCH, 128, 0, stream>>>(h1, pooled);
  head<<<1, 64, 0, stream>>>(pooled, fcw, fcb, fc2w, fc2b, out);
}
